// GraphTripleConv_44530220925731
// MI455X (gfx1250) — matrix operand order repaired and checked by reading, not compiled, not run on hardware
//
#include <hip/hip_runtime.h>
#include <hip/hip_bf16.h>

typedef __attribute__((ext_vector_type(16))) __bf16         v16bf;
typedef __attribute__((ext_vector_type(8)))  float          v8f;
typedef __attribute__((ext_vector_type(8)))  unsigned short us8;
typedef __attribute__((ext_vector_type(16))) unsigned short us16;

// builtin expects pointer-to-int4 (gcc vector_size spelling) in global/shared AS
typedef int v4i __attribute__((vector_size(16)));
typedef __attribute__((address_space(1))) v4i* gptr_v4i;
typedef __attribute__((address_space(3))) v4i* sptr_v4i;

#if defined(__has_builtin)
#if __has_builtin(__builtin_amdgcn_global_load_async_to_lds_b128) && \
    __has_builtin(__builtin_amdgcn_s_wait_asynccnt)
#define USE_ASYNC_LDS 1
#endif
#endif
#ifndef USE_ASYNC_LDS
#define USE_ASYNC_LDS 0
#endif

// ---------------------------------------------------------------- helpers

__device__ __forceinline__ unsigned short f2bf(float f) {
  unsigned int u = __builtin_bit_cast(unsigned int, f);
  u += 0x7FFFu + ((u >> 16) & 1u);          // round-to-nearest-even
  return (unsigned short)(u >> 16);
}

__global__ void zero_f32(float* __restrict__ p, int n) {
  int i = blockIdx.x * blockDim.x + threadIdx.x;
  int stride = gridDim.x * blockDim.x;
  for (; i < n; i += stride) p[i] = 0.0f;
}

__global__ void cvt_bf16(const float* __restrict__ in,
                         unsigned short* __restrict__ out, int n) {
  int i = blockIdx.x * blockDim.x + threadIdx.x;
  int stride = gridDim.x * blockDim.x;
  for (; i < n; i += stride) out[i] = f2bf(in[i]);
}

__global__ void count_edges(const int* __restrict__ edges,
                            float* __restrict__ counts, int ntri) {
  int i = blockIdx.x * blockDim.x + threadIdx.x;
  if (i < ntri) {
    atomicAdd(&counts[edges[2 * i]], 1.0f);
    atomicAdd(&counts[edges[2 * i + 1]], 1.0f);
  }
}

// pooled /= max(counts,1), convert to bf16
__global__ void normalize_pool(const float* __restrict__ pooled,
                               const float* __restrict__ counts,
                               unsigned short* __restrict__ out, int total) {
  int i = blockIdx.x * blockDim.x + threadIdx.x;
  int stride = gridDim.x * blockDim.x;
  for (; i < total; i += stride) {
    float c = counts[i >> 9];                 // D = 512
    c = c < 1.0f ? 1.0f : c;
    out[i] = f2bf(pooled[i] / c);
  }
}

// ---------------------------------------------------------------- WMMA GEMM
// C(MxN) = epilogue( A(MxK,bf16) @ B(NxK,bf16)^T )
// Block tile 128x128, K-step 32, double-buffered LDS filled with
// global_load_async_to_lds_b128 (ASYNCcnt) when available. 256 threads =
// 8 waves (4x2); each wave owns a 32x64 sub-tile = 2x4 fragments of
// v_wmma_f32_16x16x32_bf16.
//
// MODE 0: A gathered as [obj[s] | pred | obj[o]]; relu(+bias); store bf16 (h)
// MODE 1: relu(+bias); n<512 -> atomicAdd pooled[s]; n<1024 -> store newp f32;
//         else atomicAdd pooled[o]
// MODE 2: relu(+bias); store bf16 (g)
// MODE 3: relu(+bias); store f32 (out_obj  =)
// MODE 4: no relu;     out_obj += acc + bias        (obj @ P_obj^T + pb_obj)
// MODE 5: no relu;     out_pred = addend + acc + bias

template <int MODE>
__global__ __launch_bounds__(256) void gemm_bf16_wmma(
    const unsigned short* __restrict__ A,
    const unsigned short* __restrict__ B,
    const float* __restrict__ bias,
    const int* __restrict__ edges,
    const unsigned short* __restrict__ objbf,
    const unsigned short* __restrict__ predbf,
    float* __restrict__ pooled,
    float* __restrict__ outf,
    unsigned short* __restrict__ outbf,
    const float* __restrict__ addend,
    int M, int N, int K) {
  const int BM = 128, BK = 32, LDP = BK + 8;
  __shared__ unsigned short sA[2][128 * LDP];
  __shared__ unsigned short sB[2][128 * LDP];

  const int tid  = threadIdx.x;
  const int lane = tid & 31;
  const int wave = tid >> 5;
  const int wm = wave & 3;           // 4 wave-rows  (32 M each)
  const int wn = wave >> 2;          // 2 wave-cols  (64 N each)
  const int m0 = blockIdx.y * BM;
  const int n0 = blockIdx.x * 128;
  const int half = lane >> 4;        // lane group 0/1
  const int l16  = lane & 15;

  const v8f vzero = {0.f, 0.f, 0.f, 0.f, 0.f, 0.f, 0.f, 0.f};
  const us8 uzero = {0, 0, 0, 0, 0, 0, 0, 0};
  v8f acc[2][4];
#pragma unroll
  for (int i = 0; i < 2; ++i)
#pragma unroll
    for (int j = 0; j < 4; ++j) acc[i][j] = vzero;

  // stage one 128x32 A tile + 128x32 B tile into LDS buffer `buf`
  auto load_tile = [&](int kb, int buf) {
    const int k0 = kb * BK;
#pragma unroll
    for (int i = 0; i < 2; ++i) {
      int linear = tid * 2 + i;          // 0..511
      int row = linear >> 2;             // 0..127
      int chunk = linear & 3;            // 8-elem chunk within 32-K
      int kk = k0 + chunk * 8;
      int gr = m0 + row;
      const unsigned short* bsrc = B + (size_t)(n0 + row) * K + kk;
#if USE_ASYNC_LDS
      // async path: clamp OOB rows (their output rows are masked anyway)
      int grc = gr < M ? gr : (M - 1);
      const unsigned short* asrc;
      if (MODE == 0) {                   // gathered concat rows
        if (kk < 512)
          asrc = objbf + (size_t)edges[2 * grc] * 512 + kk;
        else if (kk < 1024)
          asrc = predbf + (size_t)grc * 512 + (kk - 512);
        else
          asrc = objbf + (size_t)edges[2 * grc + 1] * 512 + (kk - 1024);
      } else {
        asrc = A + (size_t)grc * K + kk;
      }
      __builtin_amdgcn_global_load_async_to_lds_b128(
          (gptr_v4i)asrc, (sptr_v4i)&sA[buf][row * LDP + chunk * 8], 0, 0);
      __builtin_amdgcn_global_load_async_to_lds_b128(
          (gptr_v4i)bsrc, (sptr_v4i)&sB[buf][row * LDP + chunk * 8], 0, 0);
#else
      us8 v = uzero;
      if (gr < M) {
        const unsigned short* src;
        if (MODE == 0) {
          if (kk < 512)
            src = objbf + (size_t)edges[2 * gr] * 512 + kk;
          else if (kk < 1024)
            src = predbf + (size_t)gr * 512 + (kk - 512);
          else
            src = objbf + (size_t)edges[2 * gr + 1] * 512 + (kk - 1024);
        } else {
          src = A + (size_t)gr * K + kk;
        }
        v = *(const us8*)src;
      }
      *(us8*)&sA[buf][row * LDP + chunk * 8] = v;
      *(us8*)&sB[buf][row * LDP + chunk * 8] = *(const us8*)bsrc;
#endif
    }
  };

  const int ksteps = K / BK;
  load_tile(0, 0);

  for (int kb = 0; kb < ksteps; ++kb) {
#if USE_ASYNC_LDS
    __builtin_amdgcn_s_wait_asynccnt(0);   // our tile-kb loads are in LDS
#endif
    __syncthreads();                       // everyone's tile-kb data visible
    if (kb + 1 < ksteps) load_tile(kb + 1, (kb + 1) & 1);  // overlap next

    const unsigned short* a = sA[kb & 1];
    const unsigned short* b = sB[kb & 1];

    // A fragment (16x32): lane group 0 holds K 0-7 & 16-23, group 1 the rest
    v16bf aF[2], bF[4];
#pragma unroll
    for (int fm = 0; fm < 2; ++fm) {
      int ml = wm * 32 + fm * 16 + l16;
      us8 lo = *(const us8*)&a[ml * LDP + half * 8];
      us8 hi = *(const us8*)&a[ml * LDP + 16 + half * 8];
      us16 av = __builtin_shufflevector(lo, hi, 0, 1, 2, 3, 4, 5, 6, 7, 8, 9,
                                        10, 11, 12, 13, 14, 15);
      aF[fm] = __builtin_bit_cast(v16bf, av);
    }
    // B fragment (32x16): lane = column n (W row), same K order as A (group 0: K 0-7 & 16-23)
#pragma unroll
    for (int fn = 0; fn < 4; ++fn) {
      int nl = wn * 64 + fn * 16 + l16;
      us8 lo = *(const us8*)&b[nl * LDP + half * 8];
      us8 hi = *(const us8*)&b[nl * LDP + 16 + half * 8];
      us16 bv = __builtin_shufflevector(lo, hi, 0, 1, 2, 3, 4, 5, 6, 7, 8, 9,
                                        10, 11, 12, 13, 14, 15);
      bF[fn] = __builtin_bit_cast(v16bf, bv);
    }
#pragma unroll
    for (int fm = 0; fm < 2; ++fm)
#pragma unroll
      for (int fn = 0; fn < 4; ++fn)
        acc[fm][fn] = __builtin_amdgcn_wmma_f32_16x16x32_bf16(
            false, aF[fm], false, bF[fn], (short)0, acc[fm][fn], false, false);
  }

  // epilogue (C layout: VGPR j -> M = fm*16 + 8*half + j, N = l16)
#pragma unroll
  for (int fm = 0; fm < 2; ++fm) {
#pragma unroll
    for (int fn = 0; fn < 4; ++fn) {
      int ng = n0 + wn * 64 + fn * 16 + l16;
      float bn = bias[ng];
#pragma unroll
      for (int j = 0; j < 8; ++j) {
        int mg = m0 + wm * 32 + fm * 16 + half * 8 + j;
        if (mg < M) {
          float v = acc[fm][fn][j];
          if (MODE <= 3) {                      // bias + relu paths
            v += bn;
            v = v > 0.0f ? v : 0.0f;
          }
          if (MODE == 0 || MODE == 2) {
            outbf[(size_t)mg * N + ng] = f2bf(v);
          } else if (MODE == 1) {
            if (ng < 512)
              atomicAdd(&pooled[(size_t)edges[2 * mg] * 512 + ng], v);
            else if (ng < 1024)
              outf[(size_t)mg * 512 + (ng - 512)] = v;
            else
              atomicAdd(&pooled[(size_t)edges[2 * mg + 1] * 512 + (ng - 1024)], v);
          } else if (MODE == 3) {
            outf[(size_t)mg * N + ng] = v;
          } else if (MODE == 4) {
            outf[(size_t)mg * N + ng] += v + bn;
          } else {  // MODE 5
            outf[(size_t)mg * N + ng] = addend[(size_t)mg * N + ng] + v + bn;
          }
        }
      }
    }
  }
}

// ---------------------------------------------------------------- launch

extern "C" void kernel_launch(void* const* d_in, const int* in_sizes, int n_in,
                              void* d_out, int out_size, void* d_ws,
                              size_t ws_size, hipStream_t stream) {
  const int NOBJ = 20000, NTRI = 100000, D = 512, H = 512;
  const size_t OBJ = (size_t)NOBJ * D;        // 10,240,000
  const size_t PRD = (size_t)NTRI * D;        // 51,200,000
  const size_t W1E = (size_t)H * (3 * D);     // 512*1536
  const size_t SQ  = (size_t)D * D;

  const float* obj   = (const float*)d_in[0];
  const float* pred  = (const float*)d_in[1];
  const int*   edges = (const int*)d_in[2];
  const float* W1 = (const float*)d_in[3];
  const float* b1 = (const float*)d_in[4];
  const float* W2 = (const float*)d_in[5];
  const float* b2 = (const float*)d_in[6];
  const float* V1 = (const float*)d_in[7];
  const float* c1 = (const float*)d_in[8];
  const float* V2 = (const float*)d_in[9];
  const float* c2 = (const float*)d_in[10];
  const float* Pobj = (const float*)d_in[11];
  const float* pbo  = (const float*)d_in[12];
  const float* Ppred = (const float*)d_in[13];
  const float* pbp   = (const float*)d_in[14];

  // workspace carve (256B aligned)
  char* wp = (char*)d_ws;
  auto take = [&](size_t bytes) -> void* {
    void* r = (void*)wp;
    wp += (bytes + 255) & ~(size_t)255;
    return r;
  };
  unsigned short* obj_bf  = (unsigned short*)take(OBJ * 2);
  unsigned short* pred_bf = (unsigned short*)take(PRD * 2);
  unsigned short* w1_bf   = (unsigned short*)take(W1E * 2);
  unsigned short* w2_bf   = (unsigned short*)take(W1E * 2);
  unsigned short* v1_bf   = (unsigned short*)take(SQ * 2);
  unsigned short* v2_bf   = (unsigned short*)take(SQ * 2);
  unsigned short* po_bf   = (unsigned short*)take(SQ * 2);
  unsigned short* pp_bf   = (unsigned short*)take(SQ * 2);
  unsigned short* h_bf    = (unsigned short*)take(PRD * 2);
  float*          newp    = (float*)take(PRD * 4);
  float*          pooled  = (float*)take(OBJ * 4);
  unsigned short* pool_bf = (unsigned short*)take(OBJ * 2);
  unsigned short* g_bf    = (unsigned short*)take(OBJ * 2);
  float*          counts  = (float*)take((size_t)NOBJ * 4);

  float* out_obj  = (float*)d_out;
  float* out_pred = (float*)d_out + OBJ;

  const int T = 256;
  auto cdiv = [](size_t a, size_t b) { return (unsigned)((a + b - 1) / b); };

  // zero accumulators
  zero_f32<<<cdiv(OBJ, T), T, 0, stream>>>(pooled, (int)OBJ);
  zero_f32<<<cdiv(NOBJ, T), T, 0, stream>>>(counts, NOBJ);

  // f32 -> bf16 conversions
  cvt_bf16<<<cdiv(OBJ, T), T, 0, stream>>>(obj, obj_bf, (int)OBJ);
  cvt_bf16<<<cdiv(PRD, T), T, 0, stream>>>(pred, pred_bf, (int)PRD);
  cvt_bf16<<<cdiv(W1E, T), T, 0, stream>>>(W1, w1_bf, (int)W1E);
  cvt_bf16<<<cdiv(W1E, T), T, 0, stream>>>(W2, w2_bf, (int)W1E);
  cvt_bf16<<<cdiv(SQ, T), T, 0, stream>>>(V1, v1_bf, (int)SQ);
  cvt_bf16<<<cdiv(SQ, T), T, 0, stream>>>(V2, v2_bf, (int)SQ);
  cvt_bf16<<<cdiv(SQ, T), T, 0, stream>>>(Pobj, po_bf, (int)SQ);
  cvt_bf16<<<cdiv(SQ, T), T, 0, stream>>>(Ppred, pp_bf, (int)SQ);

  count_edges<<<cdiv(NTRI, T), T, 0, stream>>>(edges, counts, NTRI);

  const unsigned mt_tri = cdiv(NTRI, 128);   // 782
  const unsigned mt_obj = cdiv(NOBJ, 128);   // 157

  // GEMM1: h = relu([obj[s]|pred|obj[o]] @ W1^T + b1)   (bf16 out)
  gemm_bf16_wmma<0><<<dim3(512 / 128, mt_tri), T, 0, stream>>>(
      nullptr, w1_bf, b1, edges, obj_bf, pred_bf, nullptr, nullptr, h_bf,
      nullptr, NTRI, 512, 1536);

  // GEMM2: new_t = relu(h @ W2^T + b2); scatter s/o slices, store new_p
  gemm_bf16_wmma<1><<<dim3(1536 / 128, mt_tri), T, 0, stream>>>(
      h_bf, w2_bf, b2, edges, nullptr, nullptr, pooled, newp, nullptr,
      nullptr, NTRI, 1536, 512);

  // pooled mean -> bf16
  normalize_pool<<<cdiv(OBJ, T), T, 0, stream>>>(pooled, counts, pool_bf,
                                                 (int)OBJ);

  // GEMM3: g = relu(pooled @ V1^T + c1)  (bf16 out)
  gemm_bf16_wmma<2><<<dim3(4, mt_obj), T, 0, stream>>>(
      pool_bf, v1_bf, c1, nullptr, nullptr, nullptr, nullptr, nullptr, g_bf,
      nullptr, NOBJ, 512, 512);

  // GEMM4a: out_obj = relu(g @ V2^T + c2)
  gemm_bf16_wmma<3><<<dim3(4, mt_obj), T, 0, stream>>>(
      g_bf, v2_bf, c2, nullptr, nullptr, nullptr, nullptr, out_obj, nullptr,
      nullptr, NOBJ, 512, 512);

  // GEMM4b: out_obj += obj @ P_obj^T + pb_obj
  gemm_bf16_wmma<4><<<dim3(4, mt_obj), T, 0, stream>>>(
      obj_bf, po_bf, pbo, nullptr, nullptr, nullptr, nullptr, out_obj, nullptr,
      nullptr, NOBJ, 512, 512);

  // GEMM5: out_pred = new_p + pred @ P_pred^T + pb_pred
  gemm_bf16_wmma<5><<<dim3(4, mt_tri), T, 0, stream>>>(
      pred_bf, pp_bf, pbp, nullptr, nullptr, nullptr, nullptr, out_pred,
      nullptr, newp, NTRI, 512, 512);

  (void)in_sizes; (void)n_in; (void)out_size; (void)ws_size;
}
